// MultiHeadAttentionRelPosEmb_13666585936364
// MI455X (gfx1250) — matrix operand order repaired and checked by reading, not compiled, not run on hardware
//
#include <hip/hip_runtime.h>
#include <stdint.h>

// ---------------- types ----------------
typedef __attribute__((ext_vector_type(16))) __bf16          v16bf;
typedef __attribute__((ext_vector_type(8)))  float           v8f;
typedef __attribute__((ext_vector_type(16))) unsigned short  v16u;
typedef __attribute__((ext_vector_type(8)))  unsigned short  v8u;

union V16 { v16u u; v16bf bf; v8u h[2]; };

static __device__ __forceinline__ unsigned short f2bf(float f) {
    union { __bf16 h; unsigned short u; } c;
    c.h = (__bf16)f;           // native cvt, RNE
    return c.u;
}

static __device__ __forceinline__ v8f wmma_bf16(const V16& a, const V16& b, v8f c) {
    return __builtin_amdgcn_wmma_f32_16x16x32_bf16(false, a.bf, false, b.bf,
                                                   (short)0, c, false, false);
}

// DPP16 butterfly helpers: reduce across each 16-lane row group (wave32 has
// two DPP rows, lanes 0-15 / 16-31, which matches the WMMA C-layout striping).
template <int CTRL>
static __device__ __forceinline__ float dpp_mov(float x) {
    union { float f; int i; } a, r;
    a.f = x;
    r.i = __builtin_amdgcn_update_dpp(0, a.i, CTRL, 0xF, 0xF, true);
    return r.f;
}
static __device__ __forceinline__ float row16_max(float x) {
    x = fmaxf(x, dpp_mov<0xB1>(x));   // quad_perm xor1
    x = fmaxf(x, dpp_mov<0x4E>(x));   // quad_perm xor2
    x = fmaxf(x, dpp_mov<0x141>(x));  // row_half_mirror (pairs quads)
    x = fmaxf(x, dpp_mov<0x140>(x));  // row_mirror (pairs 8-halves)
    return x;
}
static __device__ __forceinline__ float row16_sum(float x) {
    x += dpp_mov<0xB1>(x);
    x += dpp_mov<0x4E>(x);
    x += dpp_mov<0x141>(x);
    x += dpp_mov<0x140>(x);
    return x;
}

// ---------------- constants ----------------
// B=8, N=1024, D=1024, H=16, HD=64 ; GEMM: M=8192, K=1024, Nd=1024
#define GM 8192
#define GK 1024
#define GN 1024

// =====================================================================
// GEMM: Y = X @ W^T + bias.
//  XBF16: 0 = X fp32, 1 = X bf16
//  MODE : 0 = bf16 head-major [g][i][hd]      (Q, K projections)
//         1 = fp32 row-major  [M][N]          (final output)
//         2 = bf16 head-transposed [g][hd][i] (V projection)
// Block: 256 thr = 8 waves (2 M x 4 N), macro tile 64x256,
// wave tile 32x64 (8 WMMAs per K-step), K-step 32.
// =====================================================================
template <int XBF16, int MODE>
__global__ __launch_bounds__(256)
void gemm_wt_bias(const void* __restrict__ X,
                  const float* __restrict__ W, const float* __restrict__ bias,
                  void* __restrict__ Y)
{
    __shared__ __attribute__((aligned(32))) unsigned short a_lds[4 * 32 * 16]; // frag-ready A
    __shared__ __attribute__((aligned(32))) unsigned short b_lds[256 * 32];    // W rows bf16

    const int tid  = threadIdx.x;
    const int wave = tid >> 5;
    const int lane = tid & 31;
    const int wm   = wave >> 2;   // 0..1
    const int wn   = wave & 3;    // 0..3
    const int n0   = blockIdx.x * 256;
    const int m0   = blockIdx.y * 64;

    v8f acc[2][4] = {};

    const int arow = tid >> 2;          // 0..63
    const int akb  = (tid & 3) * 8;     // 0,8,16,24
    const int alt  = (arow & 15) + ((akb >> 4) << 4);
    const int aii  = akb & 8;           // 0 or 8
    const int adst = (arow >> 4) * 512 + alt * 16 + aii;
    const int khalf = (lane >> 4) << 4;

    for (int kt = 0; kt < GK; kt += 32) {
        // ---- stage A (64x32) into fragment-ready LDS layout (one v8u store) ----
        {
            const size_t base = (size_t)(m0 + arow) * GK + kt + akb;
            v8u av;
            if (XBF16) {
                av = *(const v8u*)&((const unsigned short*)X)[base];
            } else {
                const float* xf = (const float*)X;
                const float4 x0 = *(const float4*)&xf[base];
                const float4 x1 = *(const float4*)&xf[base + 4];
                av[0] = f2bf(x0.x); av[1] = f2bf(x0.y);
                av[2] = f2bf(x0.z); av[3] = f2bf(x0.w);
                av[4] = f2bf(x1.x); av[5] = f2bf(x1.y);
                av[6] = f2bf(x1.z); av[7] = f2bf(x1.w);
            }
            *(v8u*)&a_lds[adst] = av;
        }
        // ---- stage B = W rows (256x32) bf16 row-major (two v16u stores) ----
        {
            const float* wp = &W[(size_t)(n0 + tid) * GK + kt];
            #pragma unroll
            for (int half = 0; half < 2; ++half) {
                v16u bv;
                #pragma unroll
                for (int e = 0; e < 16; e += 4) {
                    const float4 w4 = *(const float4*)&wp[half * 16 + e];
                    bv[e + 0] = f2bf(w4.x); bv[e + 1] = f2bf(w4.y);
                    bv[e + 2] = f2bf(w4.z); bv[e + 3] = f2bf(w4.w);
                }
                *(v16u*)&b_lds[tid * 32 + half * 16] = bv;
            }
        }
        __syncthreads();

        V16 afr[2], bfr[4];
        #pragma unroll
        for (int mt = 0; mt < 2; ++mt)
            afr[mt].u = *(const v16u*)&a_lds[(wm * 2 + mt) * 512 + lane * 16];
        #pragma unroll
        for (int nt = 0; nt < 4; ++nt) {
            int nloc = wn * 64 + nt * 16 + (lane & 15);
            bfr[nt].u = *(const v16u*)&b_lds[nloc * 32 + khalf];
        }
        #pragma unroll
        for (int mt = 0; mt < 2; ++mt)
            #pragma unroll
            for (int nt = 0; nt < 4; ++nt)
                acc[mt][nt] = wmma_bf16(afr[mt], bfr[nt], acc[mt][nt]);
        __syncthreads();
    }

    // ---- epilogue ----
    const int mbv = (lane >> 4) << 3;
    #pragma unroll
    for (int mt = 0; mt < 2; ++mt) {
        #pragma unroll
        for (int nt = 0; nt < 4; ++nt) {
            const int n  = n0 + wn * 64 + nt * 16 + (lane & 15);
            const float bn = bias[n];
            #pragma unroll
            for (int v = 0; v < 8; ++v) {
                const int m = m0 + wm * 32 + mt * 16 + mbv + v;
                const float val = acc[mt][nt][v] + bn;
                const int bb = m >> 10, i = m & 1023;
                const int h  = n >> 6,  hd = n & 63;
                if (MODE == 0) {        // bf16 [g][i][hd]
                    ((unsigned short*)Y)[((size_t)(bb * 16 + h) << 16) + (size_t)i * 64 + hd] = f2bf(val);
                } else if (MODE == 2) { // bf16 [g][hd][i]
                    ((unsigned short*)Y)[((size_t)(bb * 16 + h) << 16) + (size_t)hd * 1024 + i] = f2bf(val);
                } else {                // fp32 [M][N]
                    ((float*)Y)[(size_t)m * GN + n] = val;
                }
            }
        }
    }
}

// =====================================================================
// Flash attention with additive rel_pos + mask.
// grid.x = g in [0,128) (fused b*H+h), grid.y = query slab of 128 rows.
// block = 256 thr = 8 waves, each wave owns 16 query rows; waves are
// fully independent (per-wave LDS P scratch only, no barriers).
// Vt is the head-transposed V: [g][hd][N].
// =====================================================================
__global__ __launch_bounds__(256)
void attn_kernel(const unsigned short* __restrict__ Qp,
                 const unsigned short* __restrict__ Kp,
                 const unsigned short* __restrict__ Vt,
                 const float* __restrict__ rel,
                 const int* __restrict__ mask,
                 unsigned short* __restrict__ Yw)
{
    __shared__ __attribute__((aligned(32))) unsigned short pl[8 * 16 * 32]; // per-wave P scratch

    const int tid  = threadIdx.x;
    const int wave = tid >> 5;
    const int lane = tid & 31;
    const int g    = blockIdx.x;            // fused b*H + h
    const int rb   = g & 7;                 // rel/mask batch = g % B
    const int bb   = g >> 4;                // b
    const int h    = g & 15;                // h
    const int qbase = blockIdx.y * 128 + wave * 16;

    const int nlane = lane & 15;
    const int mbv   = (lane >> 4) << 3;
    const int kb    = (lane >> 4) * 8;
    const int khalf = (lane >> 4) << 4;

    // ---- preload Q fragments (16 rows x 64 hd = 2 A fragments) ----
    V16 qf[2];
    #pragma unroll
    for (int kt2 = 0; kt2 < 2; ++kt2) {
        const size_t qoff = ((size_t)g * 1024 + qbase + nlane) * 64 + kt2 * 32 + khalf;
        qf[kt2].h[0] = *(const v8u*)&Qp[qoff];
        qf[kt2].h[1] = *(const v8u*)&Qp[qoff + 8];
    }

    v8f  o[4] = {};
    float rmax[8], rsum[8];
    #pragma unroll
    for (int v = 0; v < 8; ++v) { rmax[v] = -1e30f; rsum[v] = 0.0f; }

    for (int jt = 0; jt < 1024; jt += 32) {
        // deep prefetch of next tile's rel/mask rows (L2-resident stream)
        if (jt + 32 < 1024) {
            const size_t ron = ((size_t)rb * 1024 + qbase + mbv) * 1024 + jt + 32 + nlane;
            __builtin_prefetch(&rel[ron], 0, 1);
            __builtin_prefetch(&mask[ron], 0, 1);
        }

        // ---- S = Q @ K^T (16 x 32 keys) ----
        v8f s[2] = {};
        #pragma unroll
        for (int nt = 0; nt < 2; ++nt) {
            const int j = jt + nt * 16 + nlane;
            #pragma unroll
            for (int kt2 = 0; kt2 < 2; ++kt2) {
                V16 kf;
                kf.u = *(const v16u*)&Kp[((size_t)g * 1024 + j) * 64 + kt2 * 32 + khalf];
                s[nt] = wmma_bf16(qf[kt2], kf, s[nt]);
            }
        }

        // ---- logits = mask ? s/8 + rel : -2e9 ----
        float lg[2][8];
        #pragma unroll
        for (int nt = 0; nt < 2; ++nt) {
            const int j = jt + nt * 16 + nlane;
            #pragma unroll
            for (int v = 0; v < 8; ++v) {
                const size_t ro = ((size_t)rb * 1024 + qbase + mbv + v) * 1024 + j;
                lg[nt][v] = mask[ro] ? (s[nt][v] * 0.125f + rel[ro]) : -2e9f;
            }
        }

        // ---- online softmax (DPP16 butterflies across 16-lane row groups) ----
        float nm[8], scl[8], ps[8];
        #pragma unroll
        for (int v = 0; v < 8; ++v) {
            const float t = row16_max(fmaxf(lg[0][v], lg[1][v]));
            nm[v]   = fmaxf(rmax[v], t);
            scl[v]  = __expf(rmax[v] - nm[v]);
            rmax[v] = nm[v];
            ps[v]   = 0.0f;
        }
        #pragma unroll
        for (int nt = 0; nt < 2; ++nt) {
            #pragma unroll
            for (int v = 0; v < 8; ++v) {
                const float p = __expf(lg[nt][v] - nm[v]);
                ps[v] += p;
                pl[wave * 512 + (mbv + v) * 32 + nt * 16 + nlane] = f2bf(p);
            }
        }
        #pragma unroll
        for (int v = 0; v < 8; ++v)
            rsum[v] = rsum[v] * scl[v] + row16_sum(ps[v]);

        // rescale running output
        #pragma unroll
        for (int nt4 = 0; nt4 < 4; ++nt4)
            #pragma unroll
            for (int v = 0; v < 8; ++v)
                o[nt4][v] *= scl[v];

        asm volatile("" ::: "memory");  // keep P stores before fragment reload

        // ---- O += P @ V  (V B-fragments straight from transposed global) ----
        V16 pf;
        pf.h[0] = *(const v8u*)&pl[wave * 512 + nlane * 32 + khalf];
        pf.h[1] = *(const v8u*)&pl[wave * 512 + nlane * 32 + 8 + khalf];
        #pragma unroll
        for (int nt4 = 0; nt4 < 4; ++nt4) {
            V16 vf;
            vf.u = *(const v16u*)&Vt[((size_t)g << 16) + (size_t)(nt4 * 16 + nlane) * 1024 + jt + khalf];
            o[nt4] = wmma_bf16(pf, vf, o[nt4]);
        }
    }

    // ---- normalize and write bf16 [B][N][D] ----
    float inv[8];
    #pragma unroll
    for (int v = 0; v < 8; ++v) inv[v] = 1.0f / fmaxf(rsum[v], 1e-30f);
    #pragma unroll
    for (int nt4 = 0; nt4 < 4; ++nt4) {
        const int hd = nt4 * 16 + nlane;
        #pragma unroll
        for (int v = 0; v < 8; ++v) {
            const int qi = qbase + mbv + v;
            Yw[((size_t)(bb * 1024 + qi) << 10) + h * 64 + hd] = f2bf(o[nt4][v] * inv[v]);
        }
    }
}

// =====================================================================
extern "C" void kernel_launch(void* const* d_in, const int* in_sizes, int n_in,
                              void* d_out, int out_size, void* d_ws, size_t ws_size,
                              hipStream_t stream) {
    (void)in_sizes; (void)n_in; (void)out_size; (void)ws_size;
    const float* q   = (const float*)d_in[0];
    const float* k   = (const float*)d_in[1];
    const float* v   = (const float*)d_in[2];
    const float* rel = (const float*)d_in[3];
    const int*   msk = (const int*)d_in[4];
    const float* Wq  = (const float*)d_in[5];
    const float* bq  = (const float*)d_in[6];
    const float* Wk  = (const float*)d_in[7];
    const float* bk  = (const float*)d_in[8];
    const float* Wv  = (const float*)d_in[9];
    const float* bv  = (const float*)d_in[10];
    const float* Wo  = (const float*)d_in[11];
    const float* bo  = (const float*)d_in[12];

    const size_t SZ = (size_t)GM * 1024; // 8M elements per bf16 tensor
    unsigned short* Qp = (unsigned short*)d_ws;
    unsigned short* Kp = Qp + SZ;
    unsigned short* Vt = Kp + SZ;
    unsigned short* Yw = Vt + SZ;

    dim3 gGrid(GN / 256, GM / 64);   // (4, 128)
    gemm_wt_bias<0, 0><<<gGrid, 256, 0, stream>>>(q, Wq, bq, Qp);
    gemm_wt_bias<0, 0><<<gGrid, 256, 0, stream>>>(k, Wk, bk, Kp);
    gemm_wt_bias<0, 2><<<gGrid, 256, 0, stream>>>(v, Wv, bv, Vt);

    attn_kernel<<<dim3(128, 8), 256, 0, stream>>>(Qp, Kp, Vt, rel, msk, Yw);

    gemm_wt_bias<1, 1><<<gGrid, 256, 0, stream>>>(Yw, Wo, bo, (float*)d_out);
}
